// LSTM_HA_84112639525648
// MI455X (gfx1250) — matrix operand order repaired and checked by reading, not compiled, not run on hardware
//
#include <hip/hip_runtime.h>
#include <hip/hip_bf16.h>
#include <math.h>

// ---------------------------------------------------------------------------
// CDNA5 (gfx1250) LSTM + attention.  wave32, WMMA f32 <- f16 x f16.
// ---------------------------------------------------------------------------

typedef __attribute__((ext_vector_type(16))) _Float16 v16h;
typedef __attribute__((ext_vector_type(8)))  _Float16 v8h;
typedef __attribute__((ext_vector_type(8)))  float    v8f;

#define WMMA_F16(a, b, c) \
  __builtin_amdgcn_wmma_f32_16x16x32_f16(false, (a), false, (b), (short)0, (c), false, false)

// CDNA5 async global->LDS copy (ASYNCcnt-tracked), guarded so compilation
// cannot regress if this toolchain doesn't expose the builtins.
#if defined(__has_builtin)
#if __has_builtin(__builtin_amdgcn_global_load_async_to_lds_b128) && \
    __has_builtin(__builtin_amdgcn_s_wait_asynccnt)
#define USE_ASYNC_LDS 1
#endif
#endif
#ifndef USE_ASYNC_LDS
#define USE_ASYNC_LDS 0
#endif

#if USE_ASYNC_LDS
// builtin takes pointers to 16B int vectors in explicit address spaces
typedef int vint4_t __attribute__((vector_size(16)));
typedef __attribute__((address_space(1))) vint4_t gv4_t;  // global
typedef __attribute__((address_space(3))) vint4_t lv4_t;  // LDS
#endif

static __device__ __forceinline__ v16h cat8(v8h lo, v8h hi) {
  return __builtin_shufflevector(lo, hi, 0,1,2,3,4,5,6,7,8,9,10,11,12,13,14,15);
}
static __device__ __forceinline__ float sigmoidf_(float x) { return 1.0f / (1.0f + expf(-x)); }

#define BB   64
#define TT   2048
#define DD   128
#define HH   256
#define G4   1024   // 4*H

// ---------------------------------------------------------------------------
// GEMM: Y[M,N] = X[M,K] * Wt + bias,  f16 WMMA, f32 accum.
//   BT=false : W is (N,K) row-major (use W^T)  -> Y = X * W^T
//   BT=true  : W is (K,N) row-major            -> Y = X * W
// 128x128 tile per 256-thread block; K multiple of 32.
// ---------------------------------------------------------------------------
template <bool BT>
__global__ __launch_bounds__(256) void gemm_kernel(
    const float* __restrict__ X, const float* __restrict__ W,
    const float* __restrict__ b1, const float* __restrict__ b2,
    float* __restrict__ Y, int M, int K, int N)
{
  __shared__ _Float16 Xs[128][48];   // [m][k], row stride 96B (16B/32B aligned)
  __shared__ _Float16 Ws[128][48];   // [n][k]
  const int tid  = threadIdx.x;
  const int lane = tid & 31, wave = tid >> 5;
  const int half = lane >> 4, l16 = lane & 15;
  const int mbase = blockIdx.x * 128;
  const int nbase = blockIdx.y * 128;
  const int mgrp = wave & 3;   // 4 m-groups of 2 mtiles
  const int ngrp = wave >> 2;  // 2 n-groups of 4 ntiles

  v8f acc[2][4];
#pragma unroll
  for (int i = 0; i < 2; ++i)
#pragma unroll
    for (int j = 0; j < 4; ++j) acc[i][j] = (v8f){};

  for (int kc = 0; kc < K; kc += 32) {
    const bool pf = (kc + 32) < K;
    // ---- stage X tile (128 rows x 32 k), convert f32 -> f16 ----
#pragma unroll
    for (int it = 0; it < 4; ++it) {
      int row = (tid >> 3) + it * 32;
      int kq  = (tid & 7) * 4;
      const float* xs = X + (size_t)(mbase + row) * K + kc + kq;
      const float4 v = *(const float4*)xs;
      if (pf) __builtin_prefetch(xs + 32, 0, 3);   // next K-chunk
      Xs[row][kq + 0] = (_Float16)v.x; Xs[row][kq + 1] = (_Float16)v.y;
      Xs[row][kq + 2] = (_Float16)v.z; Xs[row][kq + 3] = (_Float16)v.w;
    }
    // ---- stage W tile into [n][k] layout ----
    if (!BT) {
#pragma unroll
      for (int it = 0; it < 4; ++it) {
        int row = (tid >> 3) + it * 32;
        int kq  = (tid & 7) * 4;
        const float* wsp = W + (size_t)(nbase + row) * K + kc + kq;
        const float4 v = *(const float4*)wsp;
        if (pf) __builtin_prefetch(wsp + 32, 0, 3);
        Ws[row][kq + 0] = (_Float16)v.x; Ws[row][kq + 1] = (_Float16)v.y;
        Ws[row][kq + 2] = (_Float16)v.z; Ws[row][kq + 3] = (_Float16)v.w;
      }
    } else {
#pragma unroll
      for (int it = 0; it < 4; ++it) {
        int kk = (tid >> 5) + it * 8;     // 0..31
        int nq = (tid & 31) * 4;          // 0..124
        const float* wsp = W + (size_t)(kc + kk) * N + nbase + nq;
        const float4 v = *(const float4*)wsp;
        if (pf) __builtin_prefetch(wsp + (size_t)32 * N, 0, 3);
        Ws[nq + 0][kk] = (_Float16)v.x; Ws[nq + 1][kk] = (_Float16)v.y;
        Ws[nq + 2][kk] = (_Float16)v.z; Ws[nq + 3][kk] = (_Float16)v.w;
      }
    }
    __syncthreads();

    // ---- compute: 2 A frags, 4 B frags, 8 WMMAs ----
    v16h af[2];
#pragma unroll
    for (int i = 0; i < 2; ++i) {
      int mrow = (mgrp * 2 + i) * 16 + l16;
      v8h lo = *(const v8h*)&Xs[mrow][half * 8];       // K = 0..7  (+8*half)
      v8h hi = *(const v8h*)&Xs[mrow][16 + half * 8];  // K = 16..23(+8*half)
      af[i] = cat8(lo, hi);
    }
#pragma unroll
    for (int j = 0; j < 4; ++j) {
      int col = (ngrp * 4 + j) * 16 + l16;
      v16h bf = cat8(*(const v8h*)&Ws[col][half * 8], *(const v8h*)&Ws[col][16 + half * 8]);
#pragma unroll
      for (int i = 0; i < 2; ++i) acc[i][j] = WMMA_F16(af[i], bf, acc[i][j]);
    }
    __syncthreads();
  }

  // ---- epilogue: bias add + store (C layout: row = r + 8*half, col = l16) ----
#pragma unroll
  for (int j = 0; j < 4; ++j) {
    int col = nbase + (ngrp * 4 + j) * 16 + l16;
    float bias = 0.f;
    if (b1) bias += b1[col];
    if (b2) bias += b2[col];
#pragma unroll
    for (int i = 0; i < 2; ++i) {
      int rowb = mbase + (mgrp * 2 + i) * 16 + half * 8;
#pragma unroll
      for (int r = 0; r < 8; ++r)
        Y[(size_t)(rowb + r) * N + col] = acc[i][j][r] + bias;
    }
  }
}

// ---------------------------------------------------------------------------
// LSTM scan.  grid = 16 WGs = 4 batch-groups (16 rows) x 4 hidden slices
// (64 h-dims -> 256 gate columns).  Whh slice lives in LDS pre-swizzled
// into WMMA B-fragment layout (128KB, CDNA5-only).  h state in LDS (f16),
// c state in registers.  Per step: WMMA gate GEMM -> fused gate update ->
// h-slice exchange between the 4 slice-WGs via fenced atomic counter with
// parity double buffering; h gathered back via async global->LDS copy.
// ---------------------------------------------------------------------------
__global__ __launch_bounds__(256) void lstm_scan_kernel(
    const float* __restrict__ xp,    // (B,T,4H) f32, precomputed input proj
    const float* __restrict__ Whh,   // (4H,H)   f32
    float* __restrict__ hout,        // (B,T,H)  f32
    _Float16* __restrict__ hExch,    // [2][4][16][H] f16
    int* __restrict__ cnt)           // [4] sync counters (pre-zeroed)
{
  __shared__ uint32_t BLDS[32768];      // 128 fragments x 32 lanes x 8 dwords
  __shared__ float    gLDS[16][264];    // gate pre-activations (local cols)
  __shared__ _Float16 hA[16][264];      // h state, [m][k] f16 (row = 528B)

  const int tid  = threadIdx.x;
  const int lane = tid & 31, wave = tid >> 5;
  const int half = lane >> 4, l16 = lane & 15;
  const int grp = blockIdx.x >> 2;  // batch group 0..3
  const int cg  = blockIdx.x & 3;   // hidden slice 0..3

  // local col n (0..255) <-> global gate col: 256*(n>>6) + 64*cg + (n&63)
  // ---- preload Whh slice into B-fragment layout (f32 -> f16) ----
  {
    const int lane_s = tid & 31;
    const int fbase  = tid >> 5;
#pragma unroll 1
    for (int it = 0; it < 16; ++it) {
      int frag = fbase + it * 8;                  // frag = ntL*8 + kt
      int kt = frag & 7, ntL = frag >> 3;
      int col = 256 * (ntL >> 2) + 64 * cg + (ntL & 3) * 16 + (lane_s & 15);
      int kb  = kt * 32 + (lane_s >> 4) * 8;      // B frag: K = 8*(lane/16)+e (+8 for e>=8)
      union { _Float16 h[16]; uint32_t u[8]; } pk;
#pragma unroll
      for (int e = 0; e < 16; ++e)
        pk.h[e] = (_Float16)Whh[(size_t)col * HH + kb + e + (e >= 8 ? 8 : 0)];
      uint32_t* dst = &BLDS[((size_t)frag * 32 + lane_s) * 8];
#pragma unroll
      for (int d = 0; d < 8; ++d) dst[d] = pk.u[d];
    }
  }
  for (int i = tid; i < 16 * 264; i += 256) ((_Float16*)hA)[i] = (_Float16)0.0f;
  __syncthreads();

  // update-phase ownership: (b_loc, h-dims jl+16q)
  const int b_loc = tid >> 4;
  const int jl    = tid & 15;
  float c[4] = {0.f, 0.f, 0.f, 0.f};
  const int bglob = grp * 16 + b_loc;
  const size_t xpB = (size_t)bglob * TT * G4;
  const size_t hoB = (size_t)bglob * TT * HH;
  const int ntL0 = wave * 2;

#pragma unroll 1
  for (int t = 0; t < TT; ++t) {
    // prefetch this step's xp gates (hides latency behind the GEMM)
    float xv[4][4];
#pragma unroll
    for (int q = 0; q < 4; ++q) {
      int j = 64 * cg + jl + 16 * q;
#pragma unroll
      for (int g = 0; g < 4; ++g)
        xv[q][g] = xp[xpB + (size_t)t * G4 + g * HH + j];
    }
    if (t + 1 < TT) {   // pull next step's gate lines toward the WGP
#pragma unroll
      for (int g = 0; g < 4; ++g)
        __builtin_prefetch(&xp[xpB + (size_t)(t + 1) * G4 + g * HH + 64 * cg + jl], 0, 3);
    }

    // ---- gate GEMM: 1 mtile x 2 ntiles x 8 ktiles per wave ----
    v8f acc0 = (v8f){}, acc1 = (v8f){};
#pragma unroll
    for (int kt = 0; kt < 8; ++kt) {
      v8h lo = *(const v8h*)&hA[l16][kt * 32 + half * 8];
      v8h hi = *(const v8h*)&hA[l16][kt * 32 + 16 + half * 8];
      v16h a = cat8(lo, hi);
      const v16h b0 = *(const v16h*)&BLDS[(((ntL0)     * 8 + kt) * 32 + lane) * 8];
      const v16h b1 = *(const v16h*)&BLDS[(((ntL0 + 1) * 8 + kt) * 32 + lane) * 8];
      acc0 = WMMA_F16(a, b0, acc0);
      acc1 = WMMA_F16(a, b1, acc1);
    }
#pragma unroll
    for (int r = 0; r < 8; ++r) {
      gLDS[r + 8 * half][ntL0 * 16 + l16]       = acc0[r];
      gLDS[r + 8 * half][(ntL0 + 1) * 16 + l16] = acc1[r];
    }
    __syncthreads();

    // ---- fused gate nonlinearity + state update ----
#pragma unroll
    for (int q = 0; q < 4; ++q) {
      int hl = jl + 16 * q;
      float gi = gLDS[b_loc][hl]        + xv[q][0];
      float gf = gLDS[b_loc][64 + hl]   + xv[q][1];
      float gg = gLDS[b_loc][128 + hl]  + xv[q][2];
      float go = gLDS[b_loc][192 + hl]  + xv[q][3];
      float cn = sigmoidf_(gf) * c[q] + sigmoidf_(gi) * tanhf(gg);
      float hn = sigmoidf_(go) * tanhf(cn);
      c[q] = cn;
      int j = 64 * cg + hl;
      hout[hoB + (size_t)t * HH + j] = hn;
      hExch[(((size_t)(t & 1) * 4 + grp) * 16 + b_loc) * HH + j] = (_Float16)hn;
    }
    __threadfence();     // make h-slice stores device-visible
    __syncthreads();
    if (tid == 0) {
      atomicAdd(&cnt[grp], 1);
      while (__hip_atomic_load(&cnt[grp], __ATOMIC_ACQUIRE, __HIP_MEMORY_SCOPE_AGENT)
             < 4 * (t + 1)) {
        __builtin_amdgcn_s_sleep(1);
      }
    }
    __syncthreads();
    __threadfence();     // invalidate near caches: peer slices must be fresh

    // ---- gather full h (all 4 slices) back into hA ----
    {
      const _Float16* src = &hExch[(((size_t)(t & 1) * 4 + grp) * 16) * HH];
#if USE_ASYNC_LDS
      // 16 rows x 256 f16 = 512 16B-chunks; async DMA into LDS, ASYNCcnt-tracked.
#pragma unroll
      for (int cchunk = 0; cchunk < 2; ++cchunk) {
        int chunk = tid + cchunk * 256;          // 0..511
        int row = chunk >> 5, col = chunk & 31;  // 32 chunks per 512B row
        __builtin_amdgcn_global_load_async_to_lds_b128(
            (gv4_t*)(src + (size_t)chunk * 8),
            (lv4_t*)&hA[row][col * 8], 0, 0);
      }
      __builtin_amdgcn_s_wait_asynccnt(0);
#else
      for (int i = tid; i < 16 * HH; i += 256) hA[i >> 8][i & 255] = src[i];
#endif
    }
    __syncthreads();
  }
}

// ---------------------------------------------------------------------------
// Attention tail
// ---------------------------------------------------------------------------
__global__ void zero_kernel(int* p, int n) {
  int i = blockIdx.x * blockDim.x + threadIdx.x;
  if (i < n) p[i] = 0;
}

// V[b][k] = sum_h lstm_out[b,T-1,h] * W2[h][k]
__global__ __launch_bounds__(256) void vproj_kernel(
    const float* __restrict__ lo, const float* __restrict__ W2, float* __restrict__ V)
{
  __shared__ float hT[HH];
  int b = blockIdx.x, k = threadIdx.x;
  hT[k] = lo[((size_t)b * TT + (TT - 1)) * HH + k];
  __syncthreads();
  float acc = 0.f;
  for (int h = 0; h < HH; ++h) acc += hT[h] * W2[h * HH + k];
  V[b * HH + k] = acc;
}

// scores[b,t] = sum_k aw[k] * tanh(U[bt][k] + V[b][k]);  one wave per row
__global__ __launch_bounds__(256) void score_kernel(
    const float* __restrict__ U, const float* __restrict__ V,
    const float* __restrict__ aw, float* __restrict__ scores)
{
  int wave = threadIdx.x >> 5, lane = threadIdx.x & 31;
  size_t row = (size_t)blockIdx.x * 8 + wave;
  int b = (int)(row >> 11);
  float s = 0.f;
#pragma unroll
  for (int i = 0; i < 8; ++i) {
    int k = lane + i * 32;
    s += aw[k] * tanhf(U[row * HH + k] + V[b * HH + k]);
  }
#pragma unroll
  for (int off = 16; off; off >>= 1) s += __shfl_xor(s, off, 32);
  if (lane == 0) scores[row] = s;
}

__global__ __launch_bounds__(256) void softmax_kernel(
    const float* __restrict__ scores, float* __restrict__ w)
{
  __shared__ float red[256];
  int b = blockIdx.x, tid = threadIdx.x;
  const float* s = scores + (size_t)b * TT;
  float m = -1e30f;
  for (int t = tid; t < TT; t += 256) m = fmaxf(m, s[t]);
  red[tid] = m; __syncthreads();
  for (int o = 128; o; o >>= 1) { if (tid < o) red[tid] = fmaxf(red[tid], red[tid + o]); __syncthreads(); }
  m = red[0]; __syncthreads();
  float sum = 0.f;
  for (int t = tid; t < TT; t += 256) { float e = expf(s[t] - m); w[(size_t)b * TT + t] = e; sum += e; }
  red[tid] = sum; __syncthreads();
  for (int o = 128; o; o >>= 1) { if (tid < o) red[tid] += red[tid + o]; __syncthreads(); }
  float inv = 1.0f / red[0];
  for (int t = tid; t < TT; t += 256) w[(size_t)b * TT + t] *= inv;
}

// context[b][k] = sum_t w[b,t] * lstm_out[b,t,k]
__global__ __launch_bounds__(256) void context_kernel(
    const float* __restrict__ lo, const float* __restrict__ w, float* __restrict__ ctx)
{
  __shared__ float wl[256];
  int b = blockIdx.x, k = threadIdx.x;
  float acc = 0.f;
  for (int t0 = 0; t0 < TT; t0 += 256) {
#if USE_ASYNC_LDS
    if (k < 64)   // 256 f32 = 64 16B-chunks
      __builtin_amdgcn_global_load_async_to_lds_b128(
          (gv4_t*)(w + (size_t)b * TT + t0 + k * 4),
          (lv4_t*)&wl[k * 4], 0, 0);
    __builtin_amdgcn_s_wait_asynccnt(0);
#else
    wl[k] = w[(size_t)b * TT + t0 + k];
#endif
    __syncthreads();
    for (int tt = 0; tt < 256; ++tt)
      acc += wl[tt] * lo[((size_t)b * TT + t0 + tt) * HH + k];
    __syncthreads();
  }
  ctx[b * HH + k] = acc;
}

__global__ __launch_bounds__(256) void layernorm_kernel(
    const float* __restrict__ ctx, const float* __restrict__ gamma,
    const float* __restrict__ beta, float* __restrict__ out)
{
  __shared__ float red[256];
  int b = blockIdx.x, k = threadIdx.x;
  float x = ctx[b * HH + k];
  red[k] = x; __syncthreads();
  for (int o = 128; o; o >>= 1) { if (k < o) red[k] += red[k + o]; __syncthreads(); }
  float mu = red[0] / (float)HH; __syncthreads();
  float d = x - mu;
  red[k] = d * d; __syncthreads();
  for (int o = 128; o; o >>= 1) { if (k < o) red[k] += red[k + o]; __syncthreads(); }
  float var = red[0] / (float)HH;
  out[b * HH + k] = d * rsqrtf(var + 1e-5f) * gamma[k] + beta[k];
}

// ---------------------------------------------------------------------------
extern "C" void kernel_launch(void* const* d_in, const int* in_sizes, int n_in,
                              void* d_out, int out_size, void* d_ws, size_t ws_size,
                              hipStream_t stream)
{
  (void)in_sizes; (void)n_in; (void)out_size; (void)ws_size;
  const float* x     = (const float*)d_in[0];
  const float* Wih0  = (const float*)d_in[1];
  const float* Whh0  = (const float*)d_in[2];
  const float* bih0  = (const float*)d_in[3];
  const float* bhh0  = (const float*)d_in[4];
  const float* Wih1  = (const float*)d_in[5];
  const float* Whh1  = (const float*)d_in[6];
  const float* bih1  = (const float*)d_in[7];
  const float* bhh1  = (const float*)d_in[8];
  const float* W1    = (const float*)d_in[9];
  const float* W2    = (const float*)d_in[10];
  const float* aw    = (const float*)d_in[11];
  const float* gamma = (const float*)d_in[12];
  const float* beta  = (const float*)d_in[13];
  float* out = (float*)d_out;

  const size_t M = (size_t)BB * TT;                   // 131072
  float* xp  = (float*)d_ws;                          // M * 1024 (reused as U)
  float* h1  = xp + M * (size_t)G4;                   // M * 256
  float* lo  = h1 + M * (size_t)HH;                   // M * 256
  float* V   = lo + M * (size_t)HH;                   // 64*256
  float* sc  = V + (size_t)BB * HH;                   // 64*2048
  float* wsm = sc + (size_t)BB * TT;                  // 64*2048
  float* ctx = wsm + (size_t)BB * TT;                 // 64*256
  _Float16* hx = (_Float16*)(ctx + (size_t)BB * HH);  // 2*4*16*256 f16
  int* cnt = (int*)(hx + (size_t)2 * 4 * 16 * HH);    // 4 ints
  float* U = xp;                                      // reuse xp buffer

  // layer 0: xp = x @ Wih0^T + bih0 + bhh0 ; then scan
  gemm_kernel<false><<<dim3(M / 128, G4 / 128), 256, 0, stream>>>(
      x, Wih0, bih0, bhh0, xp, (int)M, DD, G4);
  zero_kernel<<<1, 64, 0, stream>>>(cnt, 4);
  lstm_scan_kernel<<<16, 256, 0, stream>>>(xp, Whh0, h1, hx, cnt);

  // layer 1
  gemm_kernel<false><<<dim3(M / 128, G4 / 128), 256, 0, stream>>>(
      h1, Wih1, bih1, bhh1, xp, (int)M, HH, G4);
  zero_kernel<<<1, 64, 0, stream>>>(cnt, 4);
  lstm_scan_kernel<<<16, 256, 0, stream>>>(xp, Whh1, lo, hx, cnt);

  // attention
  vproj_kernel<<<BB, 256, 0, stream>>>(lo, W2, V);
  gemm_kernel<true><<<dim3(M / 128, HH / 128), 256, 0, stream>>>(
      lo, W1, nullptr, nullptr, U, (int)M, HH, HH);
  score_kernel<<<M / 8, 256, 0, stream>>>(U, V, aw, sc);
  softmax_kernel<<<BB, 256, 0, stream>>>(sc, wsm);
  context_kernel<<<BB, 256, 0, stream>>>(lo, wsm, ctx);
  layernorm_kernel<<<BB, 256, 0, stream>>>(ctx, gamma, beta, out);
}
